// ResNet_86294482911937
// MI455X (gfx1250) — matrix operand order repaired and checked by reading, not compiled, not run on hardware
//
#include <hip/hip_runtime.h>

typedef __attribute__((ext_vector_type(16))) _Float16 v16h;
typedef __attribute__((ext_vector_type(8)))  float    v8f;
typedef unsigned int uint32x4 __attribute__((ext_vector_type(4)));
typedef int          int32x4  __attribute__((ext_vector_type(4)));
typedef int          int32x8  __attribute__((ext_vector_type(8)));

#define TILE_K 32

#if defined(__has_builtin)
#if __has_builtin(__builtin_amdgcn_tensor_load_to_lds)
#define HAVE_TDM 1
#endif
#endif
#ifndef HAVE_TDM
#define HAVE_TDM 0
#endif

// ---------------------------------------------------------------------------
// TDM: DMA one 2D tile (32 rows x 16 f32 cols, row stride N) Global -> LDS.
// D# per CDNA5 ISA ch.8: group0 {count=1, lds_addr, global_addr, type=2},
// group1 {data_size=4B, pad 4 DWORDs after every 16 DWORDs -> 20-float LDS
// row stride, tensor_dim = remaining extent (OOB reads return zero),
// tile_dim0=16, tile_dim1=32, tensor_dim0_stride=N}.
// ---------------------------------------------------------------------------
__device__ __forceinline__ void tdm_load_tile(unsigned lds_off, const float* g,
                                              int rem_n, int rem_k, int strideN)
{
#if HAVE_TDM
    unsigned long long ga = (unsigned long long)(uintptr_t)g;
    uint32x4 g0;
    g0[0] = 1u;                                            // count=1 (user D#)
    g0[1] = lds_off;                                       // lds_addr
    g0[2] = (unsigned)(ga & 0xffffffffu);                  // global_addr[31:0]
    g0[3] = (unsigned)((ga >> 32) & 0x01ffffffu) | (2u << 30); // addr[56:32]|type=2
    int32x8 g1;
    g1[0] = (2 << 16) | (1 << 20) | (3 << 22) | (3 << 25); // 4B, pad_en, int=16dw, amt=4dw
    g1[1] = (rem_n & 0xffff) << 16;                        // tensor_dim0[15:0]
    g1[2] = ((unsigned)rem_n >> 16) | ((rem_k & 0xffff) << 16); // td0[31:16]|td1[15:0]
    g1[3] = ((unsigned)rem_k >> 16) | (16 << 16);          // td1[31:16]|tile_dim0=16
    g1[4] = 32;                                            // tile_dim1=32, tile_dim2=0
    g1[5] = strideN;                                       // tensor_dim0_stride[31:0]
    g1[6] = 0;
    g1[7] = 0;
    int32x4 z4 = {0, 0, 0, 0};
#if __clang_major__ >= 23
    int32x8 z8 = {0, 0, 0, 0, 0, 0, 0, 0};
    __builtin_amdgcn_tensor_load_to_lds(g0, g1, z4, z4, z8, 0);
#else
    __builtin_amdgcn_tensor_load_to_lds(g0, g1, z4, z4, 0);
#endif
#else
    (void)lds_off; (void)g; (void)rem_n; (void)rem_k; (void)strideN;
#endif
}

// ---------------------------------------------------------------------------
// 1x1-conv GEMM: out[M,N] = W[M,K] @ in[K,N], fused scale/bias/residual/relu.
// Block = 128 threads = 4 waves; each wave computes MT 16x16 tiles stacked
// in M, all sharing one B tile. B tiles are TDM double-buffered.
// ---------------------------------------------------------------------------
template <int MT>
__global__ __launch_bounds__(128) void gemm_wmma_kernel(
    const float* __restrict__ in, const float* __restrict__ wgt,
    const float* __restrict__ scale, const float* __restrict__ bias,
    const float* __restrict__ resid, float* __restrict__ out,
    int K, int N, int Cout, long wm_stride, long w_off,
    int relu, int acc)
{
    const int lane  = threadIdx.x & 31;
    const int wave  = threadIdx.x >> 5;
    const int lmod  = lane & 15;
    const int lhalf = lane >> 4;
    const int n0    = blockIdx.x * 16;
    const int mbase = (blockIdx.y * 4 + wave) * 16 * MT;

    __shared__ float BsF[2][TILE_K * 20];   // TDM pads rows to 20 floats

    v8f c[MT];
    #pragma unroll
    for (int mt = 0; mt < MT; ++mt) c[mt] = (v8f){};

#if HAVE_TDM
    if (wave == 0)
        tdm_load_tile((unsigned)(uintptr_t)&BsF[0][0], in + (size_t)0 * N + n0,
                      N - n0, K, N);
#endif

    int p = 0;
    for (int kc = 0; kc < K; kc += TILE_K, p ^= 1) {
        const bool more = (kc + TILE_K) < K;
#if HAVE_TDM
        if (wave == 0) {
            if (more) {
                tdm_load_tile((unsigned)(uintptr_t)&BsF[p ^ 1][0],
                              in + (size_t)(kc + TILE_K) * N + n0,
                              N - n0, K - (kc + TILE_K), N);
                __builtin_amdgcn_s_wait_tensorcnt(1);   // wait for current tile
            } else {
                __builtin_amdgcn_s_wait_tensorcnt(0);   // last tile: drain
            }
        }
        __syncthreads();
#else
        // synchronous fallback fill
        #pragma unroll
        for (int t = 0; t < 4; ++t) {
            int idx = threadIdx.x + t * 128;
            int kk = idx >> 4;
            int nn = idx & 15;
            float v = 0.0f;
            if (kc + kk < K && n0 + nn < N)
                v = in[(size_t)(kc + kk) * N + n0 + nn];
            BsF[p][kk * 20 + nn] = v;
        }
        __syncthreads();
#endif
        // B from LDS (f32 -> f16): n = lane%16, k = 16*(lane/16)+e
        v16h b;
        #pragma unroll
        for (int e = 0; e < 16; ++e)
            b[e] = (_Float16)BsF[p][((lhalf << 4) + e) * 20 + lmod];

        #pragma unroll
        for (int mt = 0; mt < MT; ++mt) {
            const int m_a = mbase + 16 * mt + lmod;
            v16h a;
            #pragma unroll
            for (int e = 0; e < 16; ++e) {
                int k = kc + (lhalf << 4) + e;
                float v = 0.0f;
                if (m_a < Cout && k < K)
                    v = wgt[(size_t)m_a * wm_stride + (size_t)k + w_off];
                a[e] = (_Float16)v;
            }
            c[mt] = __builtin_amdgcn_wmma_f32_16x16x32_f16(
                        false, a, false, b, (short)0, c[mt], false, false);
        }
        __syncthreads();
    }

    const int n = n0 + lmod;
    #pragma unroll
    for (int mt = 0; mt < MT; ++mt) {
        #pragma unroll
        for (int r = 0; r < 8; ++r) {
            int m = mbase + 16 * mt + r + (lhalf << 3);
            if (m < Cout && n < N) {
                float v = c[mt][r];
                if (scale) v *= scale[m];
                if (bias)  v += bias[m];
                if (resid) v += resid[(size_t)m * N + n];
                if (acc)   v += out[(size_t)m * N + n];
                if (relu)  v = v > 0.0f ? v : 0.0f;
                out[(size_t)m * N + n] = v;
            }
        }
    }
}

// ---------------------------------------------------------------------------
// Generic implicit-GEMM convolution (any KH/KW/stride/pad/dilation).
// Cooperative im2col gather into LDS; MT-way M register blocking per wave.
// ---------------------------------------------------------------------------
template <int MT>
__global__ __launch_bounds__(128) void conv_wmma_kernel(
    const float* __restrict__ in, const float* __restrict__ wgt,
    const float* __restrict__ scale, const float* __restrict__ bias,
    const float* __restrict__ resid, float* __restrict__ out,
    int Cin, int Hin, int Win, int Cout, int Hout, int Wout,
    int KH, int KW, int stride, int pad, int dil,
    long wm_stride, long w_off, int relu, int acc)
{
    const int N  = Hout * Wout;
    const int K  = Cin * KH * KW;
    const int lane  = threadIdx.x & 31;
    const int wave  = threadIdx.x >> 5;
    const int lmod  = lane & 15;
    const int lhalf = lane >> 4;
    const int n0    = blockIdx.x * 16;
    const int mbase = (blockIdx.y * 4 + wave) * 16 * MT;

    __shared__ _Float16 Bs[TILE_K * 17];

    v8f c[MT];
    #pragma unroll
    for (int mt = 0; mt < MT; ++mt) c[mt] = (v8f){};

    for (int kc = 0; kc < K; kc += TILE_K) {
        #pragma unroll
        for (int t = 0; t < 4; ++t) {
            int idx = threadIdx.x + t * 128;
            int kk = idx >> 4;
            int nn = idx & 15;
            int k = kc + kk;
            int n = n0 + nn;
            float v = 0.0f;
            if (k < K && n < N) {
                int kw = k % KW;
                int kt = k / KW;
                int kh = kt % KH;
                int ci = kt / KH;
                int ow = n % Wout;
                int oh = n / Wout;
                int ih = oh * stride - pad + kh * dil;
                int iw = ow * stride - pad + kw * dil;
                if (ih >= 0 && ih < Hin && iw >= 0 && iw < Win)
                    v = in[((size_t)ci * Hin + ih) * Win + iw];
            }
            Bs[kk * 17 + nn] = (_Float16)v;
        }
        __syncthreads();

        v16h b;
        #pragma unroll
        for (int e = 0; e < 16; ++e)
            b[e] = Bs[((lhalf << 4) + e) * 17 + lmod];

        #pragma unroll
        for (int mt = 0; mt < MT; ++mt) {
            const int m_a = mbase + 16 * mt + lmod;
            v16h a;
            #pragma unroll
            for (int e = 0; e < 16; ++e) {
                int k = kc + (lhalf << 4) + e;
                float v = 0.0f;
                if (m_a < Cout && k < K)
                    v = wgt[(size_t)m_a * wm_stride + (size_t)k + w_off];
                a[e] = (_Float16)v;
            }
            c[mt] = __builtin_amdgcn_wmma_f32_16x16x32_f16(
                        false, a, false, b, (short)0, c[mt], false, false);
        }
        __syncthreads();
    }

    const int n = n0 + lmod;
    #pragma unroll
    for (int mt = 0; mt < MT; ++mt) {
        #pragma unroll
        for (int r = 0; r < 8; ++r) {
            int m = mbase + 16 * mt + r + (lhalf << 3);
            if (m < Cout && n < N) {
                float v = c[mt][r];
                if (scale) v *= scale[m];
                if (bias)  v += bias[m];
                if (resid) v += resid[(size_t)m * N + n];
                if (acc)   v += out[(size_t)m * N + n];
                if (relu)  v = v > 0.0f ? v : 0.0f;
                out[(size_t)m * N + n] = v;
            }
        }
    }
}

// 3x3 stride-2 pad-1 max pool
__global__ void maxpool_kernel(const float* __restrict__ in, float* __restrict__ out,
                               int C, int Hin, int Win, int Hout, int Wout)
{
    int idx = blockIdx.x * blockDim.x + threadIdx.x;
    int total = C * Hout * Wout;
    if (idx >= total) return;
    int ow = idx % Wout;
    int t  = idx / Wout;
    int oh = t % Hout;
    int c  = t / Hout;
    float m = -INFINITY;
    #pragma unroll
    for (int kh = 0; kh < 3; ++kh) {
        int ih = oh * 2 - 1 + kh;
        if (ih < 0 || ih >= Hin) continue;
        #pragma unroll
        for (int kw = 0; kw < 3; ++kw) {
            int iw = ow * 2 - 1 + kw;
            if (iw < 0 || iw >= Win) continue;
            float v = in[((size_t)c * Hin + ih) * Win + iw];
            m = v > m ? v : m;
        }
    }
    out[idx] = m;
}

__global__ void bias2_kernel(const float* a, const float* b, float* o, int n)
{
    int i = blockIdx.x * blockDim.x + threadIdx.x;
    if (i < n) o[i] = a[i] + b[i];
}

__global__ void bilin_pre_kernel(const float* __restrict__ off,
                                 float* __restrict__ x0t, float* __restrict__ y0t,
                                 float* __restrict__ fxt, float* __restrict__ fyt,
                                 int H, int W)
{
    int HW = H * W;
    int idx = blockIdx.x * blockDim.x + threadIdx.x;
    if (idx >= 9 * HW) return;
    int p = idx % HW;
    int s = idx / HW;
    int ow = p % W;
    int oh = p / W;
    float dx = off[(size_t)(2 * s)     * HW + p];
    float dy = off[(size_t)(2 * s + 1) * HW + p];
    float px = dx + (float)ow;
    float py = dy + (float)oh;
    float fx0 = floorf(px);
    float fy0 = floorf(py);
    x0t[idx] = fx0;
    y0t[idx] = fy0;
    fxt[idx] = px - fx0;
    fyt[idx] = py - fy0;
}

__global__ void bilin_gather_kernel(const float* __restrict__ feat,
                                    const float* __restrict__ x0t, const float* __restrict__ y0t,
                                    const float* __restrict__ fxt, const float* __restrict__ fyt,
                                    float* __restrict__ samp,
                                    int c0, int Cc, int H, int W)
{
    int HW = H * W;
    long idx = (long)blockIdx.x * blockDim.x + threadIdx.x;
    long total = (long)Cc * 9 * HW;
    if (idx >= total) return;
    int p  = (int)(idx % HW);
    int t  = (int)(idx / HW);
    int s  = t % 9;
    int cl = t / 9;
    int c  = c0 + cl;
    int sp = s * HW + p;
    int xi = (int)x0t[sp];
    int yi = (int)y0t[sp];
    float fx = fxt[sp];
    float fy = fyt[sp];
    const float* fc = feat + (size_t)c * HW;
    auto val = [&](int xx, int yy) -> float {
        return (xx >= 0 && xx < W && yy >= 0 && yy < H) ? fc[yy * W + xx] : 0.0f;
    };
    float v = (1.0f - fx) * (1.0f - fy) * val(xi,     yi)
            +        fx  * (1.0f - fy) * val(xi + 1, yi)
            + (1.0f - fx) *        fy  * val(xi,     yi + 1)
            +        fx  *        fy  * val(xi + 1, yi + 1);
    samp[(size_t)(cl * 9 + s) * HW + p] = v;
}

// ---------------------------------------------------------------------------

extern "C" void kernel_launch(void* const* d_in, const int* in_sizes, int n_in,
                              void* d_out, int out_size, void* d_ws, size_t ws_size,
                              hipStream_t stream)
{
    int xi = -1;
    for (int i = 0; i < n_in; ++i)
        if (in_sizes[i] == 3 * 321 * 321) { xi = i; break; }
    if (xi < 0) xi = n_in - 1;
    const float* x = (const float*)d_in[xi];
    int pi = (xi == 0) ? 1 : 0;
    auto NXT = [&]() -> const float* {
        if (pi == xi) ++pi;
        const float* r = (pi < n_in) ? (const float*)d_in[pi] : nullptr;
        ++pi;
        return r;
    };

    // sorted pytree keys: bn1_b, bn1_s, conv1_w, g2d_b, g2d_w, layers,
    //                     o2d_b, o2d_w, off_b[4], off_w[4], sc_b, sc_w
    const float* bn1_b   = NXT();
    const float* bn1_s   = NXT();
    const float* conv1_w = NXT();
    const float* g2d_b   = NXT();
    (void)NXT(); // g2d_w: multiplied by 0.0 in the reference -> unused

    float* ws    = (float*)d_ws;
    float* bufA  = ws;
    float* bufB  = ws + 4194304;
    float* bufSC = ws + 8388608;
    float* bufT1 = ws + 12582912;
    float* bufT2 = ws + 13631488;
    float* off0  = ws + 14680064;
    float* off1  = off0 + 32768;
    float* bias2 = off1 + 32768;
    float* tabX0 = bias2 + 64;
    float* tabY0 = tabX0 + 16384;
    float* tabFX = tabY0 + 16384;
    float* tabFY = tabFX + 16384;
    (void)ws_size;

    auto pick_mt = [](int Cout) { return Cout >= 512 ? 4 : (Cout >= 128 ? 2 : 1); };

    auto conv = [&](const float* in, const float* w, const float* sc, const float* bi,
                    const float* res, float* out,
                    int Cin, int Hin, int Win, int Cout, int Hout, int Wout,
                    int KH, int KW, int st, int pad, int dl,
                    long wms, long wof, int relu, int acc) {
        int mt = pick_mt(Cout);
        dim3 grid((Hout * Wout + 15) / 16, (Cout + 64 * mt - 1) / (64 * mt));
        switch (mt) {
        case 4:
            conv_wmma_kernel<4><<<grid, dim3(128), 0, stream>>>(
                in, w, sc, bi, res, out, Cin, Hin, Win, Cout, Hout, Wout,
                KH, KW, st, pad, dl, wms, wof, relu, acc);
            break;
        case 2:
            conv_wmma_kernel<2><<<grid, dim3(128), 0, stream>>>(
                in, w, sc, bi, res, out, Cin, Hin, Win, Cout, Hout, Wout,
                KH, KW, st, pad, dl, wms, wof, relu, acc);
            break;
        default:
            conv_wmma_kernel<1><<<grid, dim3(128), 0, stream>>>(
                in, w, sc, bi, res, out, Cin, Hin, Win, Cout, Hout, Wout,
                KH, KW, st, pad, dl, wms, wof, relu, acc);
            break;
        }
    };
    auto gemm = [&](const float* in, const float* w, const float* sc, const float* bi,
                    const float* res, float* out,
                    int K, int N, int Cout, long wms, long wof, int relu, int acc) {
        int mt = pick_mt(Cout);
        dim3 grid((N + 15) / 16, (Cout + 64 * mt - 1) / (64 * mt));
        switch (mt) {
        case 4:
            gemm_wmma_kernel<4><<<grid, dim3(128), 0, stream>>>(
                in, w, sc, bi, res, out, K, N, Cout, wms, wof, relu, acc);
            break;
        case 2:
            gemm_wmma_kernel<2><<<grid, dim3(128), 0, stream>>>(
                in, w, sc, bi, res, out, K, N, Cout, wms, wof, relu, acc);
            break;
        default:
            gemm_wmma_kernel<1><<<grid, dim3(128), 0, stream>>>(
                in, w, sc, bi, res, out, K, N, Cout, wms, wof, relu, acc);
            break;
        }
    };

    // --- stem ---------------------------------------------------------------
    conv(x, conv1_w, bn1_s, bn1_b, nullptr, bufA,
         3, 321, 321, 64, 161, 161, 7, 7, 2, 3, 1, 147, 0, 1, 0);
    {
        int total = 64 * 81 * 81;
        maxpool_kernel<<<(total + 255) / 256, 256, 0, stream>>>(
            bufA, bufB, 64, 161, 161, 81, 81);
    }

    // --- bottleneck layers --------------------------------------------------
    struct LCfg { int planes, nblk, stride, dil; };
    const LCfg cfgs[4] = { {64, 3, 1, 1}, {128, 4, 2, 1}, {256, 6, 1, 2}, {512, 3, 1, 4} };
    int inpl = 64, H = 81, W = 81;
    float* cur = bufB;
    for (int L = 0; L < 4; ++L) {
        const LCfg cf = cfgs[L];
        const int pl = cf.planes;
        for (int bi = 0; bi < cf.nblk; ++bi) {
            const bool ds = (bi == 0);
            const float* b1 = NXT(); const float* b2 = NXT(); const float* b3 = NXT();
            const float* bd = ds ? NXT() : nullptr;
            const float* s1 = NXT(); const float* s2 = NXT(); const float* s3 = NXT();
            const float* sd = ds ? NXT() : nullptr;
            const float* w1 = NXT(); const float* w2 = NXT(); const float* w3 = NXT();
            const float* wd = ds ? NXT() : nullptr;
            const int st   = (bi == 0) ? cf.stride : 1;
            const int Hout = (st == 2) ? (H - 1) / 2 + 1 : H;
            const int Wout = (st == 2) ? (W - 1) / 2 + 1 : W;

            // t1 = relu(bn(conv1x1(x))) : 1x1 stride-1 -> TDM GEMM
            gemm(cur, w1, s1, b1, nullptr, bufT1,
                 inpl, H * W, pl, inpl, 0, 1, 0);
            // t2 = relu(bn(conv3x3(t1)))
            conv(bufT1, w2, s2, b2, nullptr, bufT2,
                 pl, H, W, pl, Hout, Wout, 3, 3, st, cf.dil, cf.dil,
                 (long)pl * 9, 0, 1, 0);
            // shortcut
            const float* scp;
            if (ds) {
                if (st == 1)
                    gemm(cur, wd, sd, bd, nullptr, bufSC,
                         inpl, H * W, pl * 4, inpl, 0, 0, 0);
                else
                    conv(cur, wd, sd, bd, nullptr, bufSC,
                         inpl, H, W, pl * 4, Hout, Wout, 1, 1, st, 0, 1,
                         inpl, 0, 0, 0);
                scp = bufSC;
            } else {
                scp = cur;
            }
            // out = relu(bn(conv1x1(t2)) + shortcut) -> TDM GEMM
            float* ob = (cur == bufA) ? bufB : bufA;
            gemm(bufT2, w3, s3, b3, scp, ob,
                 pl, Hout * Wout, pl * 4, pl, 0, 1, 0);
            cur = ob; inpl = pl * 4; H = Hout; W = Wout;
        }
    }
    const float* feature = cur;   // 2048 x 41 x 41

    const float* o2d_b = NXT();
    const float* o2d_w = NXT();
    const float* off_b[4]; for (int i = 0; i < 4; ++i) off_b[i] = NXT();
    const float* off_w[4]; for (int i = 0; i < 4; ++i) off_w[i] = NXT();
    const float* sc_b = NXT();
    const float* sc_w = NXT();

    // offset0 = sum of 4 dilated 3x3 convs (each with its bias)
    const int dils[4] = { 6, 12, 18, 24 };
    for (int i = 0; i < 4; ++i)
        conv(feature, off_w[i], nullptr, off_b[i], nullptr, off0,
             2048, 41, 41, 18, 41, 41, 3, 3, 1, dils[i], dils[i],
             (long)2048 * 9, 0, 0, i > 0);

    // gradient * 0.0 => conv(0, g2d_w, g2d_b) == broadcast(g2d_b)
    bias2_kernel<<<1, 32, 0, stream>>>(o2d_b, g2d_b, bias2, 18);
    // offset1 = offset0 + conv1x1(offset0, o2d_w) + (o2d_b + g2d_b)
    gemm(off0, o2d_w, nullptr, bias2, off0, off1,
         18, 41 * 41, 18, 18, 0, 0, 0);

    // final deformable sample_conv, chunked over 256 feature channels
    {
        int total = 9 * 41 * 41;
        bilin_pre_kernel<<<(total + 255) / 256, 256, 0, stream>>>(
            off1, tabX0, tabY0, tabFX, tabFY, 41, 41);
    }
    for (int c0 = 0; c0 < 2048; c0 += 256) {
        long total = (long)256 * 9 * 41 * 41;
        bilin_gather_kernel<<<(int)((total + 255) / 256), 256, 0, stream>>>(
            feature, tabX0, tabY0, tabFX, tabFY, bufSC, c0, 256, 41, 41);
        gemm(bufSC, sc_w, nullptr, (c0 == 0) ? sc_b : nullptr, nullptr, (float*)d_out,
             2304, 41 * 41, 21, 18432, (long)c0 * 9, 0, c0 > 0);
    }
    (void)out_size;
}
